// SRHead_82892868813358
// MI455X (gfx1250) — matrix operand order repaired and checked by reading, not compiled, not run on hardware
//
#include <hip/hip_runtime.h>
#include <hip/hip_bf16.h>
#include <math.h>

// Problem constants (B=32, S=2048, H=1024)
#define B_      32
#define S_      2048
#define H_      1024
#define MTOT    (B_ * S_)      // 65536 tokens
#define CHUNK_K 128
#define NCHUNK  (H_ / CHUNK_K) // 8

typedef __attribute__((ext_vector_type(16))) __bf16 v16bf;
typedef __attribute__((ext_vector_type(8)))  float  v8f;
typedef __attribute__((ext_vector_type(4)))  unsigned int u32x4;
typedef __attribute__((ext_vector_type(8)))  int i32x8;
typedef __attribute__((ext_vector_type(4)))  int i32x4;

#if __has_builtin(__builtin_amdgcn_tensor_load_to_lds) && \
    __has_builtin(__builtin_amdgcn_s_wait_tensorcnt)
#define HAS_TDM 1
#endif

union FragBF {
    v16bf v;
    unsigned short s[16];
    uint4 q[2];
};

__device__ __forceinline__ unsigned short f32_to_bf16(float f) {
    unsigned u = __float_as_uint(f);
    u += 0x7FFFu + ((u >> 16) & 1u);   // round-to-nearest-even
    return (unsigned short)(u >> 16);
}
__device__ __forceinline__ unsigned pack_bf16(float lo, float hi) {
    return (unsigned)f32_to_bf16(lo) | ((unsigned)f32_to_bf16(hi) << 16);
}

// LDS layout (bytes) for the main kernel:
//  f32 A double buffer: 2 x 32 rows x 144 dwords  (TDM pad: +4dw per 32dw)
#define AF32_STRIDE 144
#define AF32_BUFB   (32 * AF32_STRIDE * 4)      // 18432 per buffer
//  bf16 A tile:        32 rows x 136 halves     (stride 68 words: conflict-free)
#define ABF_STRIDE  136
#define ABF_OFF     (2 * AF32_BUFB)             // 36864
//  logits reduction:   8 waves x 32 rows x 2
#define RED_OFF     (ABF_OFF + 32 * ABF_STRIDE * 2)  // 45568
#define SMEM_TOT    (RED_OFF + 8 * 32 * 2 * 4)       // 47616

#ifdef HAS_TDM
// Issue one TDM load: 32 rows x 128 f32, row stride H, into LDS at lds_off.
// This toolchain exposes the 6-arg builtin:
//   (u32x4 g0, i32x8 g1, i32x4 g2, i32x4 g3, i32x8 g4, i32 cpol)
__device__ __forceinline__ void tdm_load_chunk(const float* gsrc, unsigned lds_off) {
    unsigned long long ga = (unsigned long long)gsrc;
    u32x4 g0;
    g0[0] = 1u;                                   // count=1, user descriptor
    g0[1] = lds_off;                              // LDS byte address
    g0[2] = (unsigned)ga;                         // global addr [31:0]
    g0[3] = (unsigned)(ga >> 32) | (2u << 30);    // addr [56:32] | type=2
    i32x8 g1;
    g1[0] = (int)((2u << 16)    // data_size: 4 bytes
                | (1u << 20)    // pad_enable
                | (4u << 22)    // pad_interval: 32 dwords
                | (3u << 25));  // pad_amount:   4 dwords
    g1[1] = (int)(1024u << 16); // tensor_dim0 = H      (bits [63:48])
    g1[2] = (int)(32u << 16);   // tensor_dim1 = 32     (bits [95:80])
    g1[3] = (int)(128u << 16);  // tile_dim0   = 128    (bits [127:112])
    g1[4] = 32;                 // tile_dim1   = 32     (bits [143:128])
    g1[5] = (int)H_;            // tensor_dim0_stride = 1024 elements
    g1[6] = 0;
    g1[7] = 0;
    i32x4 z4 = {0, 0, 0, 0};
    i32x8 z8 = {0, 0, 0, 0, 0, 0, 0, 0};
    __builtin_amdgcn_tensor_load_to_lds(g0, g1, z4, z4, z8, 0);
}
#endif

// ---------------------------------------------------------------------------
// K1: gather the formula embedding femb[b,:] (label==2, exactly one / sample)
// ---------------------------------------------------------------------------
__global__ void k_formula(const float* __restrict__ enc,
                          const int*   __restrict__ lab,
                          float*       __restrict__ femb) {
    const int b = blockIdx.x;
    __shared__ int sidx;
    if (threadIdx.x == 0) sidx = 0;
    __syncthreads();
    for (int s = threadIdx.x; s < S_; s += blockDim.x)
        if (lab[b * S_ + s] == 2) sidx = s;   // single writer per sample
    __syncthreads();
    const float* src = enc + ((size_t)b * S_ + sidx) * H_;
    for (int h = threadIdx.x; h < H_; h += blockDim.x)
        femb[b * H_ + h] = src[h];
}

// ---------------------------------------------------------------------------
// K2: W1a -> bf16, row-major N x K (B-fragment: lane = N, contiguous K)
// ---------------------------------------------------------------------------
__global__ void k_w1a_bf16(const float* __restrict__ W1,
                           unsigned short* __restrict__ Wtn) {
    const int i = blockIdx.x * blockDim.x + threadIdx.x;  // 0 .. H*H-1
    const int n = i >> 10;
    const int k = i & (H_ - 1);
    Wtn[i] = f32_to_bf16(W1[(size_t)n * (2 * H_) + k]);
}

// ---------------------------------------------------------------------------
// K3: fterm[b,h] = b1[h] + dot(femb[b,:], W1[h, H:2H])
// ---------------------------------------------------------------------------
__global__ void k_fterm(const float* __restrict__ W1,
                        const float* __restrict__ b1,
                        const float* __restrict__ femb,
                        float*       __restrict__ fterm) {
    const int h = blockIdx.x * 8 + (threadIdx.x >> 5);
    const int b = threadIdx.x & 31;
    const float* wrow = W1 + (size_t)h * (2 * H_) + H_;
    const float* fe   = femb + b * H_;
    float acc = 0.f;
    for (int f = 0; f < H_; ++f) acc = fmaf(fe[f], wrow[f], acc);
    fterm[b * H_ + h] = acc + b1[h];
}

// ---------------------------------------------------------------------------
// K4: main fused kernel. 32 token-rows x full N=1024 per workgroup, 8 waves.
//     A staged by TDM (f32) -> one-shot bf16 convert pass in LDS -> WMMA.
//     hid never leaves registers; gelu+W2 projection fused into epilogue.
// ---------------------------------------------------------------------------
__launch_bounds__(256, 1)
__global__ void k_gemm_fused(const float*          __restrict__ enc,
                             const unsigned short* __restrict__ Wtn,
                             const float*          __restrict__ fterm,
                             const float*          __restrict__ W2,
                             const float*          __restrict__ b2,
                             float*                __restrict__ logits_out) {
    __shared__ __align__(16) unsigned char smem[SMEM_TOT];
    float*          Af32 = (float*)smem;
    unsigned short* Abf  = (unsigned short*)(smem + ABF_OFF);
    float*          red  = (float*)(smem + RED_OFF);

    const int lane = threadIdx.x & 31;
    const int wave = threadIdx.x >> 5;
    const int m0   = blockIdx.x * 32;
    const int b    = m0 >> 11;              // m0 / S_
    const int r16  = lane & 15;
    const int hiK  = lane >> 4;
    const int nbase = wave * 128;

    const float* gA = enc + (size_t)m0 * H_;

    v8f acc[2][8];
    #pragma unroll
    for (int i = 0; i < 2; ++i)
        #pragma unroll
        for (int j = 0; j < 8; ++j)
            #pragma unroll
            for (int e = 0; e < 8; ++e) acc[i][j][e] = 0.f;

#ifdef HAS_TDM
    const unsigned lds_base = (unsigned)(unsigned long long)(smem);
    if (wave == 0) {
        tdm_load_chunk(gA, lds_base);               // chunk 0 -> buffer 0
        __builtin_amdgcn_s_wait_tensorcnt(0);
    }
    __syncthreads();
#else
    {   // fallback: cooperative copy of chunk 0 into f32 buffer 0
        #pragma unroll
        for (int i = 0; i < 4; ++i) {
            const int q   = threadIdx.x + 256 * i;  // float4 index
            const int row = q >> 5;
            const int kq  = (q & 31) * 4;
            const float4 f = *(const float4*)(enc + (size_t)(m0 + row) * H_ + kq);
            *(float4*)&Af32[row * AF32_STRIDE + kq + 4 * (kq >> 5)] = f;
        }
        __syncthreads();
    }
#endif

    for (int c = 0; c < NCHUNK; ++c) {
        const int cb = c & 1;
#ifdef HAS_TDM
        if (wave == 0 && c + 1 < NCHUNK)            // overlap DMA of next chunk
            tdm_load_chunk(gA + (c + 1) * CHUNK_K,
                           lds_base + (unsigned)(((c + 1) & 1) * AF32_BUFB));
#endif
        // one-shot convert pass: f32 buffer cb -> bf16 tile
        const float* src = Af32 + cb * (AF32_BUFB / 4);
        #pragma unroll
        for (int i = 0; i < 4; ++i) {
            const int q   = threadIdx.x + 256 * i;
            const int row = q >> 5;
            const int kq  = (q & 31) * 4;
            const float4 f = *(const float4*)&src[row * AF32_STRIDE + kq + 4 * (kq >> 5)];
            uint2 pk;
            pk.x = pack_bf16(f.x, f.y);
            pk.y = pack_bf16(f.z, f.w);
            *(uint2*)&Abf[row * ABF_STRIDE + kq] = pk;
        }
        __syncthreads();

        // 4 WMMA k-steps over this 128-wide chunk
        #pragma unroll
        for (int ks = 0; ks < CHUNK_K; ks += 32) {
            const int kc0 = ks + hiK * 8;           // A layout: lane>=16 -> K+8/K+24
            FragBF a0, a1;
            a0.q[0] = *(const uint4*)&Abf[r16 * ABF_STRIDE + kc0];
            a0.q[1] = *(const uint4*)&Abf[r16 * ABF_STRIDE + kc0 + 16];
            a1.q[0] = *(const uint4*)&Abf[(16 + r16) * ABF_STRIDE + kc0];
            a1.q[1] = *(const uint4*)&Abf[(16 + r16) * ABF_STRIDE + kc0 + 16];
            const int kg = c * CHUNK_K + ks + hiK * 8;   // B layout: lane>=16 -> K+8/K+24
            #pragma unroll
            for (int nt = 0; nt < 8; ++nt) {
                const unsigned short* brow =
                    Wtn + (size_t)(nbase + nt * 16 + r16) * H_ + kg;
                FragBF bf;
                bf.q[0] = *(const uint4*)(brow);
                bf.q[1] = *(const uint4*)(brow + 16);
                acc[0][nt] = __builtin_amdgcn_wmma_f32_16x16x32_bf16(
                    false, a0.v, false, bf.v, (short)0, acc[0][nt], false, false);
                acc[1][nt] = __builtin_amdgcn_wmma_f32_16x16x32_bf16(
                    false, a1.v, false, bf.v, (short)0, acc[1][nt], false, false);
            }
        }

#ifdef HAS_TDM
        if (wave == 0 && c + 1 < NCHUNK)
            __builtin_amdgcn_s_wait_tensorcnt(0);   // next chunk landed
        __syncthreads();                            // publish + WAR protection
#else
        if (c + 1 < NCHUNK) {
            __syncthreads();
            #pragma unroll
            for (int i = 0; i < 4; ++i) {
                const int q   = threadIdx.x + 256 * i;
                const int row = q >> 5;
                const int kq  = (q & 31) * 4;
                const float4 f = *(const float4*)(enc + (size_t)(m0 + row) * H_ +
                                                  (c + 1) * CHUNK_K + kq);
                *(float4*)&Af32[((c + 1) & 1) * (AF32_BUFB / 4) +
                                row * AF32_STRIDE + kq + 4 * (kq >> 5)] = f;
            }
        }
        __syncthreads();
#endif
    }

    // ---- fused epilogue: gelu(acc + fterm) projected by W2 (2 x H) ----
    float plg[2][8][2];
    #pragma unroll
    for (int mt = 0; mt < 2; ++mt)
        #pragma unroll
        for (int r = 0; r < 8; ++r) { plg[mt][r][0] = 0.f; plg[mt][r][1] = 0.f; }

    #pragma unroll
    for (int nt = 0; nt < 8; ++nt) {
        const int h     = nbase + nt * 16 + r16;    // C/D layout: col = lane&15
        const float ft  = fterm[b * H_ + h];
        const float w20 = W2[h];
        const float w21 = W2[H_ + h];
        #pragma unroll
        for (int mt = 0; mt < 2; ++mt)
            #pragma unroll
            for (int r = 0; r < 8; ++r) {           // VGPR r -> row r (+8 if lane>=16)
                float x = acc[mt][nt][r] + ft;
                float g = 0.5f * x * (1.f + erff(x * 0.70710678118654752f));
                plg[mt][r][0] = fmaf(g, w20, plg[mt][r][0]);
                plg[mt][r][1] = fmaf(g, w21, plg[mt][r][1]);
            }
    }

    #pragma unroll
    for (int mt = 0; mt < 2; ++mt)
        #pragma unroll
        for (int r = 0; r < 8; ++r)
            #pragma unroll
            for (int c = 0; c < 2; ++c) {
                float v = plg[mt][r][c];
                v += __shfl_xor(v, 1, 32);          // reduce over 16 columns
                v += __shfl_xor(v, 2, 32);
                v += __shfl_xor(v, 4, 32);
                v += __shfl_xor(v, 8, 32);
                if (r16 == 0)
                    red[(wave * 32 + mt * 16 + hiK * 8 + r) * 2 + c] = v;
            }
    __syncthreads();
    if (threadIdx.x < 64) {
        const int row = threadIdx.x >> 1;
        const int cc  = threadIdx.x & 1;
        float s = b2[cc];
        #pragma unroll
        for (int w = 0; w < 8; ++w) s += red[(w * 32 + row) * 2 + cc];
        logits_out[(size_t)(m0 + row) * 2 + cc] = s;
    }
}

// ---------------------------------------------------------------------------
// K5: per-sample CE over header tokens + header_mask output.
// ---------------------------------------------------------------------------
__global__ void k_losspart(const int*   __restrict__ lab,
                           const float* __restrict__ logits,
                           float*       __restrict__ hm_out,
                           float*       __restrict__ sl,
                           float*       __restrict__ valid) {
    const int b = blockIdx.x;
    float lsum = 0.f, cnt = 0.f;
    for (int s = threadIdx.x; s < S_; s += blockDim.x) {
        const int l = lab[b * S_ + s];
        const bool hm = (l == 0) || (l == 1);
        hm_out[b * S_ + s] = hm ? 1.f : 0.f;
        if (hm) {
            const float l0 = logits[((size_t)b * S_ + s) * 2 + 0];
            const float l1 = logits[((size_t)b * S_ + s) * 2 + 1];
            const float mx = fmaxf(l0, l1);
            const float lse = mx + logf(expf(l0 - mx) + expf(l1 - mx));
            lsum += lse - (l ? l1 : l0);
            cnt  += 1.f;
        }
    }
    __shared__ float ss[256], sc[256];
    ss[threadIdx.x] = lsum; sc[threadIdx.x] = cnt;
    __syncthreads();
    for (int st = 128; st > 0; st >>= 1) {
        if ((int)threadIdx.x < st) {
            ss[threadIdx.x] += ss[threadIdx.x + st];
            sc[threadIdx.x] += sc[threadIdx.x + st];
        }
        __syncthreads();
    }
    if (threadIdx.x == 0) {
        const float c = sc[0];
        sl[b]    = ss[0] / fmaxf(c, 1.f);
        valid[b] = (c > 0.f) ? 1.f : 0.f;
    }
}

// K6: final masked mean over the 32 samples.
__global__ void k_lossfinal(const float* __restrict__ sl,
                            const float* __restrict__ valid,
                            float*       __restrict__ out) {
    const int t = threadIdx.x;  // 32 threads
    float a = sl[t] * valid[t];
    float v = valid[t];
    #pragma unroll
    for (int m = 16; m > 0; m >>= 1) {
        a += __shfl_xor(a, m, 32);
        v += __shfl_xor(v, m, 32);
    }
    if (t == 0) out[0] = a / (v + 1e-6f);
}

// ---------------------------------------------------------------------------
extern "C" void kernel_launch(void* const* d_in, const int* in_sizes, int n_in,
                              void* d_out, int out_size, void* d_ws, size_t ws_size,
                              hipStream_t stream) {
    const float* enc = (const float*)d_in[0];
    const int*   lab = (const int*)d_in[1];
    const float* W1  = (const float*)d_in[2];
    const float* b1  = (const float*)d_in[3];
    const float* W2  = (const float*)d_in[4];
    const float* b2  = (const float*)d_in[5];
    float* out = (float*)d_out;

    // workspace layout
    char* ws = (char*)d_ws;
    float*          femb  = (float*)ws;                        // 128 KB
    unsigned short* Wtn   = (unsigned short*)(ws + 131072);    // 2 MB
    float*          fterm = (float*)(ws + 131072 + 2097152);   // 128 KB
    float*          sl    = fterm + B_ * H_;
    float*          valid = sl + B_;

    // d_out layout: [0]=loss, [1 .. 1+2*MTOT)=logits, [1+2*MTOT ..)=header_mask
    float* logits = out + 1;
    float* hmask  = out + 1 + 2 * MTOT;

    k_formula   <<<B_,              256, 0, stream>>>(enc, lab, femb);
    k_w1a_bf16  <<<(H_ * H_) / 256, 256, 0, stream>>>(W1, Wtn);
    k_fterm     <<<H_ / 8,          256, 0, stream>>>(W1, b1, femb, fterm);
    k_gemm_fused<<<MTOT / 32,       256, 0, stream>>>(enc, Wtn, fterm, W2, b2, logits);
    k_losspart  <<<B_,              256, 0, stream>>>(lab, logits, hmask, sl, valid);
    k_lossfinal <<<1,                32, 0, stream>>>(sl, valid, out);
}
